// SpatioAttention_66589172957645
// MI455X (gfx1250) — matrix operand order repaired and checked by reading, not compiled, not run on hardware
//
#include <hip/hip_runtime.h>
#include <hip/hip_bf16.h>

#define DIMF 256
#define HEADS 8
#define HID 2048
#define BT 64
#define LSEQ 256
#define SK 264   // padded LDS row stride (halves) for attention tiles

typedef __attribute__((ext_vector_type(16))) _Float16 v16h;
typedef __attribute__((ext_vector_type(8)))  _Float16 v8h;
typedef __attribute__((ext_vector_type(8)))  float    v8f;
typedef __attribute__((ext_vector_type(4)))  unsigned int u32x4;
typedef __attribute__((ext_vector_type(8)))  int      i32x8;
typedef __attribute__((ext_vector_type(4)))  int      i32x4;
typedef int b128i __attribute__((vector_size(16)));   // matches async-LDS builtin pointee

// ---- CDNA5 async / TDM feature probes (fallback to sync copies if absent) ----
#if defined(__has_builtin)
#if __has_builtin(__builtin_amdgcn_global_load_async_to_lds_b128) && \
    __has_builtin(__builtin_amdgcn_s_wait_asynccnt)
#define USE_ASYNC 1
#endif
#if __has_builtin(__builtin_amdgcn_tensor_load_to_lds) && \
    __has_builtin(__builtin_amdgcn_s_wait_tensorcnt)
#define USE_TDM 1
#endif
#endif
#ifndef USE_ASYNC
#define USE_ASYNC 0
#endif
#ifndef USE_TDM
#define USE_TDM 0
#endif

union AFrag { v16h v; v8h h[2]; };

// A-matrix 16x32 f16 fragment from row-major [m][k] buffer (row stride ld halves).
// ISA layout: lanes 0-15 hold M=lane, K {0..7,16..23}; lanes 16-31 hold M=lane-16, K {8..15,24..31}.
__device__ __forceinline__ v16h frag_a(const _Float16* tile, int ld) {
  const int lane = threadIdx.x & 31;
  const int r = lane & 15;
  const int k0 = (lane < 16) ? 0 : 8;
  AFrag a;
  a.h[0] = *(const v8h*)(tile + r * ld + k0);
  a.h[1] = *(const v8h*)(tile + r * ld + k0 + 16);
  return a.v;
}

// B-matrix 32x16 f16 fragment from an n-major [n][k] buffer (row stride ld halves).
// ISA layout: lanes 0-15 hold N=lane, K {0..7,16..23}; lanes 16-31 hold N=lane-16, K {8..15,24..31}.
__device__ __forceinline__ v16h frag_b(const _Float16* tileT, int ld) {
  const int lane = threadIdx.x & 31;
  const int n = lane & 15;
  const int k0 = (lane < 16) ? 0 : 8;
  AFrag b;
  b.h[0] = *(const v8h*)(tileT + n * ld + k0);
  b.h[1] = *(const v8h*)(tileT + n * ld + k0 + 16);
  return b.v;
}

__device__ __forceinline__ v8f wmma16(v16h a, v16h b, v8f c) {
  return __builtin_amdgcn_wmma_f32_16x16x32_f16(false, a, false, b, (short)0, c,
                                                false, false);
}

// 16-byte global->LDS copy: async data mover when available, else VGPR round-trip.
__device__ __forceinline__ void cp16_g2l(const void* g, void* l) {
#if USE_ASYNC
  __builtin_amdgcn_global_load_async_to_lds_b128(
      (__attribute__((address_space(1))) b128i*)g,
      (__attribute__((address_space(3))) b128i*)l, 0, 0);
#else
  *(v8h*)l = *(const v8h*)g;
#endif
}
__device__ __forceinline__ void async_fence() {
#if USE_ASYNC
  __builtin_amdgcn_s_wait_asynccnt(0);
#endif
}

__device__ __forceinline__ float mish_f(float x) {
  float sp = (x > 20.f) ? x : __logf(1.f + __expf(x));
  return x * tanhf(sp);
}

// ---------------------------------------------------------------------------
// Kernel 0: one-time weight reformat to f16, transposed to n-major [n][k]
// WqT/WkT/WvT: [2048][256]; WoT: [256][2048]. grid(2048), block(256)
// ---------------------------------------------------------------------------
__global__ __launch_bounds__(256) void wtrans_kernel(
    const float* __restrict__ Wq, const float* __restrict__ Wk,
    const float* __restrict__ Wv, const float* __restrict__ Wo,
    _Float16* __restrict__ WqT, _Float16* __restrict__ WkT,
    _Float16* __restrict__ WvT, _Float16* __restrict__ WoT) {
  const int idx = blockIdx.x * 256 + threadIdx.x;  // 0 .. 524287
  {  // W[256][2048] -> WT[2048][256]
    const int n = idx >> 8;     // 0..2047
    const int k = idx & 255;    // 0..255
    const int s = k * HID + n;
    WqT[idx] = (_Float16)Wq[s];
    WkT[idx] = (_Float16)Wk[s];
    WvT[idx] = (_Float16)Wv[s];
  }
  {  // Wo[2048][256] -> WoT[256][2048]
    const int n = idx >> 11;    // 0..255
    const int k = idx & 2047;   // 0..2047
    WoT[idx] = (_Float16)Wo[k * DIMF + n];
  }
}

// ---------------------------------------------------------------------------
// Kernel 1: m = mish(t_embed + reward_embed);  biasX[bt] = m @ Wx_t + bx
// grid(64), block(256)
// ---------------------------------------------------------------------------
__global__ __launch_bounds__(256) void bias_kernel(
    const float* __restrict__ t_embed, const float* __restrict__ r_embed,
    const float* __restrict__ Wq_t, const float* __restrict__ bq_t,
    const float* __restrict__ Wk_t, const float* __restrict__ bk_t,
    const float* __restrict__ Wv_t, const float* __restrict__ bv_t,
    float* __restrict__ biasQ, float* __restrict__ biasK, float* __restrict__ biasV) {
  __shared__ float m[DIMF];
  const int bt = blockIdx.x;
  const int tid = threadIdx.x;
  m[tid] = mish_f(t_embed[bt * DIMF + tid] + r_embed[bt * DIMF + tid]);
  __syncthreads();
  const float* Wt[3] = {Wq_t, Wk_t, Wv_t};
  const float* bb[3] = {bq_t, bk_t, bv_t};
  float* outp[3] = {biasQ, biasK, biasV};
  for (int mat = 0; mat < 3; ++mat) {
    for (int c = tid; c < HID; c += 256) {
      float s = bb[mat][c];
      const float* w = Wt[mat] + c;
      for (int f = 0; f < DIMF; ++f) s += m[f] * w[f * HID];
      outp[mat][bt * HID + c] = s;
    }
  }
}

// ---------------------------------------------------------------------------
// Kernel 2: Q/K/V projection GEMM: Y[bt][l][2048] = x[bt] @ W + bias[bt], f16 out
// grid(16 nblk, 3*2 mat/mblk, 64 bt), block(256) = 8 waves; block tile 128x128
// ---------------------------------------------------------------------------
__global__ __launch_bounds__(256) void proj_kernel(
    const float* __restrict__ x,
    const _Float16* __restrict__ WqT, const _Float16* __restrict__ WkT,
    const _Float16* __restrict__ WvT,
    const float* __restrict__ biasQ, const float* __restrict__ biasK,
    const float* __restrict__ biasV,
    _Float16* __restrict__ Qws, _Float16* __restrict__ Kws, _Float16* __restrict__ Vws) {
  const int nblk = blockIdx.x;
  const int mat  = blockIdx.y >> 1;
  const int mblk = blockIdx.y & 1;
  const int bt   = blockIdx.z;
  const _Float16* WT = (mat == 0) ? WqT : ((mat == 1) ? WkT : WvT);
  const float* bias  = (mat == 0) ? biasQ : ((mat == 1) ? biasK : biasV);
  _Float16*    Y     = (mat == 0) ? Qws : ((mat == 1) ? Kws : Vws);

  __shared__ _Float16 As[128 * 32];  // [m][k]
  __shared__ _Float16 Bs[128 * 32];  // [n][k]

  const int tid = threadIdx.x;
  const int wid = tid >> 5;
  const int wm = wid >> 1;  // 0..3
  const int wn = wid & 1;   // 0..1
  const int m0 = mblk * 128;
  const int n0 = nblk * 128;

  v8f acc[2][4] = {};
  const float* xb = x + (size_t)(bt * LSEQ + m0) * DIMF;

  for (int k0 = 0; k0 < DIMF; k0 += 32) {
    {  // stage A 128x32, f32 -> f16 (needs conversion: VALU path)
      const int row = tid >> 1;
      const int c0 = (tid & 1) * 16;
      const float* src = xb + row * DIMF + k0 + c0;
      _Float16* dst = As + row * 32 + c0;
      if (k0 + 32 < DIMF) __builtin_prefetch(src + 32, 0, 1);
#pragma unroll
      for (int i = 0; i < 16; ++i) dst[i] = (_Float16)src[i];
    }
    {  // stage B n-major 128x32: straight f16 copy (async data mover path)
      const int n = tid >> 1;
      const int c0 = (tid & 1) * 16;
      const _Float16* src = WT + (size_t)(n0 + n) * DIMF + k0 + c0;
      _Float16* dst = Bs + n * 32 + c0;
      cp16_g2l(src, dst);
      cp16_g2l(src + 8, dst + 8);
    }
    async_fence();
    __syncthreads();
    v16h af[2], bf[4];
#pragma unroll
    for (int i = 0; i < 2; ++i) af[i] = frag_a(As + (wm * 32 + i * 16) * 32, 32);
#pragma unroll
    for (int j = 0; j < 4; ++j) bf[j] = frag_b(Bs + (wn * 64 + j * 16) * 32, 32);
#pragma unroll
    for (int i = 0; i < 2; ++i)
#pragma unroll
      for (int j = 0; j < 4; ++j) acc[i][j] = wmma16(af[i], bf[j], acc[i][j]);
    __syncthreads();
  }

  const int lane = tid & 31;
  const int cl = lane & 15;
  const int rhi = (lane < 16) ? 0 : 8;
#pragma unroll
  for (int i = 0; i < 2; ++i)
#pragma unroll
    for (int j = 0; j < 4; ++j) {
      const int col = n0 + wn * 64 + j * 16 + cl;
      const float bv = bias[bt * HID + col];
      const int rb = m0 + wm * 32 + i * 16 + rhi;
#pragma unroll
      for (int e = 0; e < 8; ++e) {
        Y[(size_t)(bt * LSEQ + rb + e) * HID + col] = (_Float16)(acc[i][j][e] + bv);
      }
    }
}

// ---------------------------------------------------------------------------
// Kernel 3: attention per (head, bt). Whole K head + V^T head resident in LDS
// (CDNA5: 320KB per WGP). K head staged by the Tensor Data Mover with HW
// LDS padding (128 DWORD interval, 4 DWORD pad -> row stride SK=264 halves).
// grid(8 heads, 64 bt), block(256) = 8 waves; ~315KB dynamic LDS.
// ---------------------------------------------------------------------------
__global__ __launch_bounds__(256) void attn_kernel(
    const _Float16* __restrict__ Qws, const _Float16* __restrict__ Kws,
    const _Float16* __restrict__ Vws, _Float16* __restrict__ AOws) {
  extern __shared__ char smem[];
  _Float16* Ksh = (_Float16*)smem;                   // [256][SK]  K[j][d]
  _Float16* Vt  = (_Float16*)(smem + 135168);        // [256][SK]  V^T[d][j]
  float*    sc  = (float*)(smem + 270336);           // [32][256] scores
  _Float16* at  = (_Float16*)(smem + 303104);        // [32][SK]  attn (f16)
  float*    red = (float*)(smem + 320000);           // [32][8] reductions

  const int h = blockIdx.x;
  const int bt = blockIdx.y;
  const int tid = threadIdx.x;
  const int wid = tid >> 5;
  const int lane = tid & 31;

  const _Float16* Qg = Qws + (size_t)(bt * LSEQ) * HID + h * DIMF;
  const _Float16* Kg = Kws + (size_t)(bt * LSEQ) * HID + h * DIMF;
  const _Float16* Vg = Vws + (size_t)(bt * LSEQ) * HID + h * DIMF;

#if USE_TDM
  if (wid == 0) {  // one TDM descriptor: 256x256 f16 tile, LDS pad 4dw/128dw
    const unsigned long long ga = (unsigned long long)(uintptr_t)Kg;
    const unsigned lds_off =
        (unsigned)(uintptr_t)(__attribute__((address_space(3))) char*)(void*)Ksh;
    u32x4 g0;
    g0[0] = 1u;                                   // count=1, user descriptor
    g0[1] = lds_off;                              // lds_addr (bytes)
    g0[2] = (unsigned)(ga & 0xffffffffu);         // global_addr[31:0]
    g0[3] = (unsigned)((ga >> 32) & 0x1ffffffu)   // global_addr[56:32]
            | 0x80000000u;                        // type=2 (image)
    i32x8 g1;
    g1[0] = (int)((1u << 16)        // data_size = 2 bytes
                | (1u << 20)        // pad_enable
                | (6u << 22)        // pad_interval: 128 DWORDs
                | (3u << 25));      // pad_amount: 4 DWORDs
    g1[1] = (int)(2048u << 16);     // tensor_dim0 = 2048 (row length, elems)
    g1[2] = (int)(256u << 16);      // tensor_dim0 hi=0 | tensor_dim1 = 256
    g1[3] = (int)(256u << 16);      // tensor_dim1 hi=0 | tile_dim0 = 256
    g1[4] = 256;                    // tile_dim1 = 256, tile_dim2 = 0
    g1[5] = 2048;                   // tensor_dim0_stride = 2048
    g1[6] = 0;
    g1[7] = 0;
    i32x4 gz;
    gz[0] = 0; gz[1] = 0; gz[2] = 0; gz[3] = 0;
#if __clang_major__ >= 23
    i32x8 gz8;
    gz8[0] = 0; gz8[1] = 0; gz8[2] = 0; gz8[3] = 0;
    gz8[4] = 0; gz8[5] = 0; gz8[6] = 0; gz8[7] = 0;
    __builtin_amdgcn_tensor_load_to_lds(g0, g1, gz, gz, gz8, 0);
#else
    __builtin_amdgcn_tensor_load_to_lds(g0, g1, gz, gz, 0);
#endif
    __builtin_amdgcn_s_wait_tensorcnt((short)0);
  }
#else
  {  // stage K row-major: thread = key row j
    const _Float16* src = Kg + (size_t)tid * HID;
    _Float16* dst = Ksh + tid * SK;
#pragma unroll
    for (int d = 0; d < DIMF; d += 8) cp16_g2l(src + d, dst + d);
  }
#endif
  {  // stage V transposed: thread = dim d, coalesced global reads over j
    const int d = tid;
    for (int j = 0; j < LSEQ; ++j) Vt[d * SK + j] = Vg[(size_t)j * HID + d];
  }
  async_fence();
  __syncthreads();

  const float scale = 0.0625f;  // 1/sqrt(256)
  const int mt = wid & 1;         // which 16-row M tile of the 32-row chunk
  const int ntb = (wid >> 1) * 4; // 4 consecutive 16-col N tiles per wave

  for (int qc = 0; qc < 8; ++qc) {
    const int q0 = qc * 32;
    // ---- scores = Q K^T * scale ----
    v8f acc[4] = {};
    for (int k0 = 0; k0 < DIMF; k0 += 32) {
      v16h af = frag_a(Qg + (size_t)(q0 + mt * 16) * HID + k0, HID);
#pragma unroll
      for (int j = 0; j < 4; ++j) {
        v16h bf = frag_b(Ksh + (ntb + j) * 16 * SK + k0, SK);
        acc[j] = wmma16(af, bf, acc[j]);
      }
    }
    {
      const int cl = lane & 15;
      const int rhi = (lane < 16) ? 0 : 8;
#pragma unroll
      for (int j = 0; j < 4; ++j) {
        const int col = (ntb + j) * 16 + cl;
#pragma unroll
        for (int e = 0; e < 8; ++e)
          sc[(mt * 16 + rhi + e) * 256 + col] = acc[j][e] * scale;
      }
    }
    __syncthreads();
    // ---- softmax over 256 cols, 8 threads per row ----
    {
      const int r = tid >> 3;
      const int seg = tid & 7;
      float mx = -1e30f;
#pragma unroll
      for (int c = 0; c < 32; ++c) mx = fmaxf(mx, sc[r * 256 + seg * 32 + c]);
      red[r * 8 + seg] = mx;
      __syncthreads();
      float rm = red[r * 8];
#pragma unroll
      for (int s = 1; s < 8; ++s) rm = fmaxf(rm, red[r * 8 + s]);
      float ex[32];
      float sum = 0.f;
#pragma unroll
      for (int c = 0; c < 32; ++c) {
        float e = __expf(sc[r * 256 + seg * 32 + c] - rm);
        ex[c] = e;
        sum += e;
      }
      __syncthreads();
      red[r * 8 + seg] = sum;
      __syncthreads();
      float tot = 0.f;
#pragma unroll
      for (int s = 0; s < 8; ++s) tot += red[r * 8 + s];
      const float inv = 1.f / tot;
#pragma unroll
      for (int c = 0; c < 32; ++c)
        at[r * SK + seg * 32 + c] = (_Float16)(ex[c] * inv);
    }
    __syncthreads();
    // ---- out = attn @ V ----
    v8f oacc[4] = {};
    for (int k0 = 0; k0 < LSEQ; k0 += 32) {
      v16h af = frag_a(at + (mt * 16) * SK + k0, SK);
#pragma unroll
      for (int j = 0; j < 4; ++j) {
        v16h bf = frag_b(Vt + (ntb + j) * 16 * SK + k0, SK);
        oacc[j] = wmma16(af, bf, oacc[j]);
      }
    }
    {
      const int cl = lane & 15;
      const int rhi = (lane < 16) ? 0 : 8;
#pragma unroll
      for (int j = 0; j < 4; ++j) {
        const int d = (ntb + j) * 16 + cl;
#pragma unroll
        for (int e = 0; e < 8; ++e) {
          const int r = q0 + mt * 16 + rhi + e;
          AOws[(size_t)(bt * LSEQ + r) * HID + h * DIMF + d] = (_Float16)oacc[j][e];
        }
      }
    }
    __syncthreads();  // protect sc/at reuse next chunk
  }
}

// ---------------------------------------------------------------------------
// Kernel 4: out = AO[16384][2048] @ Wo[2048][256] + bo, f32 to d_out
// grid(2 nblk, 128 mblk), block(256); block tile 128x128, K loop 2048
// ---------------------------------------------------------------------------
__global__ __launch_bounds__(256) void outproj_kernel(
    const _Float16* __restrict__ AO, const _Float16* __restrict__ WoT,
    const float* __restrict__ bo, float* __restrict__ out) {
  __shared__ _Float16 As[128 * 32];
  __shared__ _Float16 Bs[128 * 32];
  const int nblk = blockIdx.x;
  const int mblk = blockIdx.y;
  const int tid = threadIdx.x;
  const int wid = tid >> 5;
  const int wm = wid >> 1;
  const int wn = wid & 1;
  const int m0 = mblk * 128;
  const int n0 = nblk * 128;

  v8f acc[2][4] = {};

  for (int k0 = 0; k0 < HID; k0 += 32) {
    {  // stage A: straight f16 copy (async path)
      const int row = tid >> 1;
      const int c0 = (tid & 1) * 16;
      const _Float16* src = AO + (size_t)(m0 + row) * HID + k0 + c0;
      _Float16* dst = As + row * 32 + c0;
      cp16_g2l(src, dst);
      cp16_g2l(src + 8, dst + 8);
    }
    {  // stage B n-major from WoT: straight f16 copy (async path)
      const int n = tid >> 1;
      const int c0 = (tid & 1) * 16;
      const _Float16* src = WoT + (size_t)(n0 + n) * HID + k0 + c0;
      _Float16* dst = Bs + n * 32 + c0;
      cp16_g2l(src, dst);
      cp16_g2l(src + 8, dst + 8);
    }
    async_fence();
    __syncthreads();
    v16h af[2], bf[4];
#pragma unroll
    for (int i = 0; i < 2; ++i) af[i] = frag_a(As + (wm * 32 + i * 16) * 32, 32);
#pragma unroll
    for (int j = 0; j < 4; ++j) bf[j] = frag_b(Bs + (wn * 64 + j * 16) * 32, 32);
#pragma unroll
    for (int i = 0; i < 2; ++i)
#pragma unroll
      for (int j = 0; j < 4; ++j) acc[i][j] = wmma16(af[i], bf[j], acc[i][j]);
    __syncthreads();
  }

  const int lane = tid & 31;
  const int cl = lane & 15;
  const int rhi = (lane < 16) ? 0 : 8;
#pragma unroll
  for (int i = 0; i < 2; ++i)
#pragma unroll
    for (int j = 0; j < 4; ++j) {
      const int col = n0 + wn * 64 + j * 16 + cl;
      const float bv = bo[col];
      const int rb = m0 + wm * 32 + i * 16 + rhi;
#pragma unroll
      for (int e = 0; e < 8; ++e)
        out[(size_t)(rb + e) * DIMF + col] = acc[i][j][e] + bv;
    }
}

// ---------------------------------------------------------------------------
extern "C" void kernel_launch(void* const* d_in, const int* in_sizes, int n_in,
                              void* d_out, int out_size, void* d_ws, size_t ws_size,
                              hipStream_t stream) {
  (void)in_sizes; (void)n_in; (void)out_size; (void)ws_size;
  const float* qkv_x   = (const float*)d_in[0];
  const float* t_embed = (const float*)d_in[1];
  const float* r_embed = (const float*)d_in[2];
  const float* Wq      = (const float*)d_in[3];
  const float* Wk      = (const float*)d_in[4];
  const float* Wv      = (const float*)d_in[5];
  const float* Wq_t    = (const float*)d_in[6];
  const float* bq_t    = (const float*)d_in[7];
  const float* Wk_t    = (const float*)d_in[8];
  const float* bk_t    = (const float*)d_in[9];
  const float* Wv_t    = (const float*)d_in[10];
  const float* bv_t    = (const float*)d_in[11];
  const float* Wo      = (const float*)d_in[12];
  const float* bo      = (const float*)d_in[13];

  // workspace layout
  char* ws = (char*)d_ws;
  float* biasQ = (float*)ws;                 // 3 x 64*2048 f32
  float* biasK = biasQ + BT * HID;
  float* biasV = biasK + BT * HID;
  _Float16* WqT = (_Float16*)(ws + (size_t)3 * BT * HID * sizeof(float));
  const size_t wT_elems = (size_t)DIMF * HID;       // 524288
  _Float16* WkT = WqT + wT_elems;
  _Float16* WvT = WkT + wT_elems;
  _Float16* WoT = WvT + wT_elems;
  _Float16* Qws = WoT + wT_elems;
  const size_t qkv_elems = (size_t)BT * LSEQ * HID;
  _Float16* Kws = Qws + qkv_elems;
  _Float16* Vws = Kws + qkv_elems;
  _Float16* AOws = Vws + qkv_elems;

  wtrans_kernel<<<dim3(2048), dim3(256), 0, stream>>>(Wq, Wk, Wv, Wo,
                                                      WqT, WkT, WvT, WoT);

  bias_kernel<<<dim3(BT), dim3(256), 0, stream>>>(
      t_embed, r_embed, Wq_t, bq_t, Wk_t, bk_t, Wv_t, bv_t, biasQ, biasK, biasV);

  proj_kernel<<<dim3(16, 6, BT), dim3(256), 0, stream>>>(
      qkv_x, WqT, WkT, WvT, biasQ, biasK, biasV, Qws, Kws, Vws);

  attn_kernel<<<dim3(HEADS, BT), dim3(256), 321024, stream>>>(Qws, Kws, Vws, AOws);

  outproj_kernel<<<dim3(2, 128), dim3(256), 0, stream>>>(AOws, WoT, bo, (float*)d_out);
}
